// DotProductAttention_90263032693333
// MI455X (gfx1250) — matrix operand order repaired and checked by reading, not compiled, not run on hardware
//
#include <hip/hip_runtime.h>
#include <stdint.h>

typedef _Float16 half4 __attribute__((ext_vector_type(4)));
typedef _Float16 half8 __attribute__((ext_vector_type(8)));
typedef _Float16 v16h  __attribute__((ext_vector_type(16)));
typedef float    v8f   __attribute__((ext_vector_type(8)));

#define SEQ    4096
#define DHEAD  64
#define BM     64     // query rows per block (4 waves x 16)
#define BN     32     // S (key) tile == K-dim of the P*V WMMA
#define QP     72     // LDS pitch (halves) for Q tile
#define KP     72     // LDS pitch (halves) for K tile  (32 dwords + 4 pad = TDM pad 4/3)
#define VP     40     // LDS pitch (halves) for transposed V tile (16 dwords + 4 pad = TDM pad 3/3)
#define PP     40     // LDS pitch (halves) for per-wave P scratch
#define NEGBIG (-1.0e9f)

// ---- TDM availability: device pass only (host pass lacks amdgcn builtins) ----
#if defined(__AMDGCN__)
#  if __has_builtin(__builtin_amdgcn_tensor_load_to_lds) && __has_builtin(__builtin_amdgcn_s_wait_tensorcnt)
#    define USE_TDM 1
#  endif
#endif
#ifndef USE_TDM
#  define USE_TDM 0
#endif

__device__ __forceinline__ float fast_exp2(float x) {
  return __builtin_amdgcn_exp2f(x);  // v_exp_f32 (hardware exp2)
}

__device__ __forceinline__ v16h pack16(half8 lo, half8 hi) {
  v16h r;
#pragma unroll
  for (int i = 0; i < 8; ++i) { r[i] = lo[i]; r[i + 8] = hi[i]; }
  return r;
}

#if USE_TDM
typedef unsigned int u32x4 __attribute__((ext_vector_type(4)));
typedef int          i32x4 __attribute__((ext_vector_type(4)));
typedef int          i32x8 __attribute__((ext_vector_type(8)));

// 2-D f16 tile load via Tensor Data Mover. Descriptor per CDNA5 ISA D# groups 0/1.
// tensor dims == tile dims (tile never OOB); pad fields recreate the LDS pitch.
__device__ __forceinline__ void tdm_load_tile_2d(unsigned lds_off, const void* gptr,
                                                 unsigned tile0, unsigned tile1,
                                                 unsigned stride0,
                                                 unsigned pad_i, unsigned pad_a) {
  const unsigned long long ga = (unsigned long long)(uintptr_t)gptr;
  u32x4 g0;
  g0[0] = 1u;                                                  // count=1, user descriptor
  g0[1] = lds_off;                                             // lds_addr (bytes)
  g0[2] = (unsigned)ga;                                        // global_addr[31:0]
  g0[3] = (unsigned)((ga >> 32) & 0x01ffffffull) | (2u << 30); // addr[56:32] | type=2
  i32x8 g1;
  g1[0] = (int)((1u << 16) | (1u << 20) | (pad_i << 22) | (pad_a << 25)); // data_size=2B, pad_enable
  g1[1] = (int)((tile0 & 0xffffu) << 16);                      // tensor_dim0[15:0]
  g1[2] = (int)(((tile0 >> 16) & 0xffffu) | ((tile1 & 0xffffu) << 16)); // dim0 hi | dim1 lo
  g1[3] = (int)(((tile1 >> 16) & 0xffffu) | ((tile0 & 0xffffu) << 16)); // dim1 hi | tile_dim0
  g1[4] = (int)(tile1 & 0xffffu);                              // tile_dim1 (tile_dim2=0)
  g1[5] = (int)stride0;                                        // tensor_dim0_stride[31:0]
  g1[6] = 0;
  g1[7] = 0;
  i32x4 z4 = {0, 0, 0, 0};
#if __clang_major__ >= 23
  i32x8 z8 = {0, 0, 0, 0, 0, 0, 0, 0};
  __builtin_amdgcn_tensor_load_to_lds(g0, g1, z4, z4, z8, 0);
#else
  __builtin_amdgcn_tensor_load_to_lds(g0, g1, z4, z4, 0);
#endif
}
#endif  // USE_TDM

// =====================  prepass kernels (fill d_ws)  =====================

// fp32 -> f16, contiguous (for K)
__global__ __launch_bounds__(256) void cvt_f16_kernel(const float* __restrict__ X,
                                                      _Float16* __restrict__ Y, int total8) {
  const int i = blockIdx.x * 256 + threadIdx.x;
  if (i >= total8) return;
  const float4 a = ((const float4*)X)[2 * i];
  const float4 b = ((const float4*)X)[2 * i + 1];
  half8 h;
  h[0] = (_Float16)a.x; h[1] = (_Float16)a.y; h[2] = (_Float16)a.z; h[3] = (_Float16)a.w;
  h[4] = (_Float16)b.x; h[5] = (_Float16)b.y; h[6] = (_Float16)b.z; h[7] = (_Float16)b.w;
  ((half8*)Y)[i] = h;
}

// V[h][s][v] fp32 -> Vt[h][v][s] f16 (64x64 tiles through LDS)
__global__ __launch_bounds__(256) void tr_v_kernel(const float* __restrict__ V,
                                                   _Float16* __restrict__ Vo) {
  __shared__ _Float16 t[64 * 66];
  const int tid = threadIdx.x;
  const int head = blockIdx.y;
  const int s0 = blockIdx.x * 64;
  const float* __restrict__ Vh = V + ((size_t)head * SEQ + s0) * DHEAD;
#pragma unroll 2
  for (int i = tid; i < 64 * 16; i += 256) {
    const int r = i >> 4, c4 = (i & 15) * 4;
    const float4 v4 = *(const float4*)(Vh + (size_t)r * DHEAD + c4);
    t[r * 66 + c4 + 0] = (_Float16)v4.x;
    t[r * 66 + c4 + 1] = (_Float16)v4.y;
    t[r * 66 + c4 + 2] = (_Float16)v4.z;
    t[r * 66 + c4 + 3] = (_Float16)v4.w;
  }
  __syncthreads();
  _Float16* __restrict__ Voh = Vo + (size_t)head * DHEAD * SEQ;
#pragma unroll 2
  for (int i = tid; i < 64 * 8; i += 256) {
    const int v = i >> 3, c8 = (i & 7) * 8;
    half8 h;
#pragma unroll
    for (int j = 0; j < 8; ++j) h[j] = t[(c8 + j) * 66 + v];
    *(half8*)(&Voh[(size_t)v * SEQ + s0 + c8]) = h;
  }
}

// bool bytes -> bit-packed mask words (bit k of word i == byte 32*i+k nonzero)
__global__ __launch_bounds__(256) void pack_mask_kernel(const unsigned char* __restrict__ M,
                                                        unsigned* __restrict__ Mp, int totalW) {
  const int i = blockIdx.x * 256 + threadIdx.x;
  if (i >= totalW) return;
  const uint4 a = ((const uint4*)M)[2 * i];
  const uint4 b = ((const uint4*)M)[2 * i + 1];
  const unsigned w[8] = {a.x, a.y, a.z, a.w, b.x, b.y, b.z, b.w};
  unsigned bits = 0;
#pragma unroll
  for (int j = 0; j < 8; ++j)
#pragma unroll
    for (int k = 0; k < 4; ++k)
      bits |= ((w[j] >> (8 * k)) & 0xffu) ? (1u << (4 * j + k)) : 0u;
  Mp[i] = bits;
}

// =====================  primary attention kernel  =====================
// Computes S^T = K_tile * Q^T so each lane owns one query's scores in-register:
// softmax max/sum are in-lane + one xor-16 shuffle; mask is 1 b32 word/lane/tile.

__global__ __launch_bounds__(128) void fa_tdm_kernel(
    const float* __restrict__ Q, const _Float16* __restrict__ Kf,
    const _Float16* __restrict__ Vtg, const unsigned* __restrict__ Mp,
    float* __restrict__ O) {
  __shared__ _Float16 Qs[BM * QP];
  __shared__ _Float16 Ks[2 * BN * KP];      // double-buffered K tile
  __shared__ _Float16 Vt[2 * DHEAD * VP];   // double-buffered transposed V tile
  __shared__ _Float16 Ps[4 * 16 * PP];      // per-wave P scratch, [q][s] order
  __shared__ float    Al[4 * 16];           // per-wave alpha / l broadcast

  const int tid  = threadIdx.x;
  const int wave = tid >> 5;
  const int lane = tid & 31;
  const int l16  = lane & 15;
  const int hi   = lane >> 4;
  const int head = blockIdx.y;
  const int qBase = blockIdx.x * BM;

  const float* __restrict__ Qh = Q + (size_t)head * SEQ * DHEAD;
  const _Float16* __restrict__ Kh = Kf + (size_t)head * SEQ * DHEAD;
  const _Float16* __restrict__ Vh = Vtg + (size_t)head * DHEAD * SEQ;
  float* __restrict__ Oh = O + (size_t)head * SEQ * DHEAD;

  // ---- stage Q tile, folding (1/sqrt(d)) * log2(e) so scores are exp2-domain ----
  const float qscale = 0.125f * 1.4426950408889634f;
#pragma unroll 2
  for (int i = tid; i < BM * (DHEAD / 4); i += 128) {
    const int r = i >> 4, c4 = (i & 15) * 4;
    const float4 q4 = *(const float4*)(Qh + (size_t)(qBase + r) * DHEAD + c4);
    half4 h;
    h[0] = (_Float16)(q4.x * qscale); h[1] = (_Float16)(q4.y * qscale);
    h[2] = (_Float16)(q4.z * qscale); h[3] = (_Float16)(q4.w * qscale);
    *(half4*)(&Qs[r * QP + c4]) = h;
  }
  __syncthreads();

  // ---- B-fragments of Q^T (column q = this lane), resident in registers ----
  const int qcol = wave * 16 + l16;  // query owned by this lane
  v16h bQ[2];
#pragma unroll
  for (int dc = 0; dc < 2; ++dc) {
    const half8 lo = *(const half8*)(&Qs[qcol * QP + dc * 32 + hi * 16]);
    const half8 hh = *(const half8*)(&Qs[qcol * QP + dc * 32 + hi * 16 + 8]);
    bQ[dc] = pack16(lo, hh);
  }

  float mstate = -3.0e38f, lstate = 0.0f;  // per-lane (per-query) softmax state
  v8f acc[4];
#pragma unroll
  for (int t = 0; t < 4; ++t) acc[t] = (v8f){0.f, 0.f, 0.f, 0.f, 0.f, 0.f, 0.f, 0.f};

  _Float16* __restrict__ Pw = &Ps[wave * 16 * PP];
  float* __restrict__ Alw = &Al[wave * 16];
  const unsigned* __restrict__ mrowp = Mp + (size_t)head * SEQ * (SEQ / 32)
                                          + (size_t)(qBase + qcol) * (SEQ / 32);

  auto stage = [&](int buf, int sB) {
#if USE_TDM
    if (wave == 0) {
      tdm_load_tile_2d((unsigned)(uintptr_t)(&Ks[buf * BN * KP]),
                       Kh + (size_t)sB * DHEAD, DHEAD, BN, DHEAD, 4u, 3u);
      tdm_load_tile_2d((unsigned)(uintptr_t)(&Vt[buf * DHEAD * VP]),
                       Vh + sB, BN, DHEAD, SEQ, 3u, 3u);
    }
#else
    _Float16* kd = &Ks[buf * BN * KP];
#pragma unroll
    for (int i = tid; i < BN * 8; i += 128) {
      const int r = i >> 3, c8 = (i & 7) * 8;
      *(half8*)(&kd[r * KP + c8]) = *(const half8*)(Kh + (size_t)(sB + r) * DHEAD + c8);
    }
    _Float16* vd = &Vt[buf * DHEAD * VP];
#pragma unroll
    for (int i = tid; i < DHEAD * 4; i += 128) {
      const int v = i >> 2, c8 = (i & 3) * 8;
      *(half8*)(&vd[v * VP + c8]) = *(const half8*)(Vh + (size_t)v * SEQ + sB + c8);
    }
#endif
  };

  stage(0, 0);

  const int NT = SEQ / BN;
  for (int it = 0; it < NT; ++it) {
    const int cur = it & 1;
    const int sB = it * BN;
    if (it + 1 < NT) stage(cur ^ 1, sB + BN);
#if USE_TDM
    if (wave == 0) {
      if (it + 1 < NT) __builtin_amdgcn_s_wait_tensorcnt(2);
      else             __builtin_amdgcn_s_wait_tensorcnt(0);
    }
#endif
    __syncthreads();

    const _Float16* __restrict__ KsB = &Ks[cur * BN * KP];
    const _Float16* __restrict__ VtB = &Vt[cur * DHEAD * VP];

    // ---- S^T(32x16) = K_tile(32x64) * Q^T(64x16); VGPR r <-> s row, lane <-> q ----
    v8f c[2];
#pragma unroll
    for (int st = 0; st < 2; ++st) {
      v8f cz = (v8f){0.f, 0.f, 0.f, 0.f, 0.f, 0.f, 0.f, 0.f};
#pragma unroll
      for (int dc = 0; dc < 2; ++dc) {
        const int srow = st * 16 + l16;  // A-fragment row (s)
        const half8 a0 = *(const half8*)(&KsB[srow * KP + dc * 32 + hi * 16]);
        const half8 a1 = *(const half8*)(&KsB[srow * KP + dc * 32 + hi * 16 + 8]);
        const v16h aK = pack16(a0, a1);
        cz = __builtin_amdgcn_wmma_f32_16x16x32_f16(false, aK, false, bQ[dc],
                                                    (short)0, cz, false, false);
      }
      c[st] = cz;
    }

    // ---- mask (one word per lane) + in-lane max over this lane's 16 s-values ----
    const unsigned mw = mrowp[it];
    float x[2][8];
    float mx = -3.0e38f;
#pragma unroll
    for (int st = 0; st < 2; ++st)
#pragma unroll
      for (int r = 0; r < 8; ++r) {
        const int sl = st * 16 + r + 8 * hi;
        const float v = ((mw >> sl) & 1u) ? c[st][r] : NEGBIG;
        x[st][r] = v;
        mx = fmaxf(mx, v);
      }
    mx = fmaxf(mx, __shfl_xor(mx, 16, 32));  // combine lane pair (other half of s)

    const float mNew = fmaxf(mstate, mx);
    const float alpha = fast_exp2(mstate - mNew);
    float sum = 0.0f;
#pragma unroll
    for (int st = 0; st < 2; ++st)
#pragma unroll
      for (int r = 0; r < 8; ++r) {
        const float p = fast_exp2(x[st][r] - mNew);
        x[st][r] = p;
        sum += p;
      }
    sum += __shfl_xor(sum, 16, 32);
    lstate = lstate * alpha + sum;
    mstate = mNew;

    // ---- stash P in [q][s] order (A-layout friendly) + alpha broadcast ----
#pragma unroll
    for (int st = 0; st < 2; ++st)
#pragma unroll
      for (int r = 0; r < 8; ++r)
        Pw[l16 * PP + st * 16 + r + 8 * hi] = (_Float16)x[st][r];
    if (hi == 0) Alw[l16] = alpha;
    asm volatile("s_wait_dscnt 0" ::: "memory");

    // ---- rescale accumulators (alpha per O-row read via LDS broadcast) ----
    float aRow[8];
#pragma unroll
    for (int r = 0; r < 8; ++r) aRow[r] = Alw[r + 8 * hi];
#pragma unroll
    for (int t = 0; t < 4; ++t)
#pragma unroll
      for (int r = 0; r < 8; ++r) acc[t][r] *= aRow[r];

    // ---- O += P(16x32) * V(32x64) ----
    {
      const half8 plo = *(const half8*)(&Pw[l16 * PP + hi * 8]);
      const half8 phi = *(const half8*)(&Pw[l16 * PP + hi * 8 + 16]);
      const v16h pA = pack16(plo, phi);
#pragma unroll
      for (int nt = 0; nt < 4; ++nt) {
        const half8 b0 = *(const half8*)(&VtB[(nt * 16 + l16) * VP + hi * 8]);
        const half8 b1 = *(const half8*)(&VtB[(nt * 16 + l16) * VP + hi * 8 + 16]);
        const v16h vB = pack16(b0, b1);
        acc[nt] = __builtin_amdgcn_wmma_f32_16x16x32_f16(false, pA, false, vB,
                                                         (short)0, acc[nt], false, false);
      }
    }
    __syncthreads();
  }

  // ---- epilogue: broadcast l per row, normalize, store fp32 ----
  if (hi == 0) Alw[l16] = lstate;
  asm volatile("s_wait_dscnt 0" ::: "memory");
  float lRow[8];
#pragma unroll
  for (int r = 0; r < 8; ++r) lRow[r] = Alw[r + 8 * hi];
#pragma unroll
  for (int t = 0; t < 4; ++t)
#pragma unroll
    for (int r = 0; r < 8; ++r) {
      const int qRow = qBase + wave * 16 + r + 8 * hi;
      Oh[(size_t)qRow * DHEAD + t * 16 + l16] = acc[t][r] / lRow[r];
    }
}

// =====================  fallback (small workspace): self-contained kernel  =====================

__global__ __launch_bounds__(128) void fa_fwd_kernel(
    const float* __restrict__ Q, const float* __restrict__ K,
    const float* __restrict__ V, const unsigned char* __restrict__ M,
    float* __restrict__ O) {
  __shared__ _Float16 Qs[BM * QP];
  __shared__ _Float16 Ks[BN * KP];
  __shared__ _Float16 Vt[DHEAD * VP];
  __shared__ _Float16 Ps[4 * 16 * PP];

  const int tid  = threadIdx.x;
  const int wave = tid >> 5;
  const int lane = tid & 31;
  const int l16  = lane & 15;
  const int hi   = lane >> 4;
  const int head = blockIdx.y;
  const int qBase = blockIdx.x * BM;

  const float* __restrict__ Qh = Q + (size_t)head * SEQ * DHEAD;
  const float* __restrict__ Kh = K + (size_t)head * SEQ * DHEAD;
  const float* __restrict__ Vh = V + (size_t)head * SEQ * DHEAD;
  const unsigned char* __restrict__ Mh = M + (size_t)head * SEQ * SEQ;
  float* __restrict__ Oh = O + (size_t)head * SEQ * DHEAD;

#pragma unroll 2
  for (int i = tid; i < BM * (DHEAD / 4); i += 128) {
    const int r = i >> 4, c4 = (i & 15) * 4;
    const float4 q4 = *(const float4*)(Qh + (size_t)(qBase + r) * DHEAD + c4);
    half4 h;
    h[0] = (_Float16)q4.x; h[1] = (_Float16)q4.y;
    h[2] = (_Float16)q4.z; h[3] = (_Float16)q4.w;
    *(half4*)(&Qs[r * QP + c4]) = h;
  }
  __syncthreads();

  v16h aF[2];
#pragma unroll
  for (int dc = 0; dc < 2; ++dc) {
    const int row = wave * 16 + l16;
    const half8 lo = *(const half8*)(&Qs[row * QP + dc * 32 + hi * 8]);
    const half8 hh = *(const half8*)(&Qs[row * QP + dc * 32 + hi * 8 + 16]);
    aF[dc] = pack16(lo, hh);
  }

  float mrow[8], lrow[8];
  v8f acc[4];
#pragma unroll
  for (int r = 0; r < 8; ++r) { mrow[r] = -3.0e38f; lrow[r] = 0.0f; }
#pragma unroll
  for (int t = 0; t < 4; ++t) acc[t] = (v8f){0.f, 0.f, 0.f, 0.f, 0.f, 0.f, 0.f, 0.f};

  _Float16* __restrict__ Pw = &Ps[wave * 16 * PP];
  const float scale = 0.125f;

  for (int sB = 0; sB < SEQ; sB += BN) {
    __syncthreads();
#pragma unroll
    for (int i = tid; i < BN * (DHEAD / 4); i += 128) {
      const int r = i >> 4, c4 = (i & 15) * 4;
      const float4 k4 = *(const float4*)(Kh + (size_t)(sB + r) * DHEAD + c4);
      half4 h;
      h[0] = (_Float16)k4.x; h[1] = (_Float16)k4.y;
      h[2] = (_Float16)k4.z; h[3] = (_Float16)k4.w;
      *(half4*)(&Ks[r * KP + c4]) = h;
    }
#pragma unroll
    for (int i = tid; i < BN * (DHEAD / 4); i += 128) {
      const int r = i >> 4, c4 = (i & 15) * 4;
      const float4 v4 = *(const float4*)(Vh + (size_t)(sB + r) * DHEAD + c4);
      Vt[(c4 + 0) * VP + r] = (_Float16)v4.x;
      Vt[(c4 + 1) * VP + r] = (_Float16)v4.y;
      Vt[(c4 + 2) * VP + r] = (_Float16)v4.z;
      Vt[(c4 + 3) * VP + r] = (_Float16)v4.w;
    }
    __syncthreads();

    v8f c[2];
#pragma unroll
    for (int nt = 0; nt < 2; ++nt) {
      v8f cz = (v8f){0.f, 0.f, 0.f, 0.f, 0.f, 0.f, 0.f, 0.f};
#pragma unroll
      for (int dc = 0; dc < 2; ++dc) {
        const int srow = nt * 16 + l16;
        const half8 b0 = *(const half8*)(&Ks[srow * KP + dc * 32 + hi * 8]);
        const half8 b1 = *(const half8*)(&Ks[srow * KP + dc * 32 + hi * 8 + 16]);
        const v16h bF = pack16(b0, b1);
        cz = __builtin_amdgcn_wmma_f32_16x16x32_f16(false, aF[dc], false, bF,
                                                    (short)0, cz, false, false);
      }
      c[nt] = cz;
    }

    float pv0[8], pv1[8], rmax[8];
#pragma unroll
    for (int r = 0; r < 8; ++r) {
      const int qRow = qBase + wave * 16 + r + 8 * hi;
      const size_t mb = (size_t)qRow * SEQ + sB;
      const float v0 = Mh[mb + l16]      ? c[0][r] * scale : NEGBIG;
      const float v1 = Mh[mb + 16 + l16] ? c[1][r] * scale : NEGBIG;
      pv0[r] = v0; pv1[r] = v1;
      float rm = fmaxf(v0, v1);
      rm = fmaxf(rm, __shfl_xor(rm, 1, 32));
      rm = fmaxf(rm, __shfl_xor(rm, 2, 32));
      rm = fmaxf(rm, __shfl_xor(rm, 4, 32));
      rm = fmaxf(rm, __shfl_xor(rm, 8, 32));
      rmax[r] = rm;
    }

#pragma unroll
    for (int r = 0; r < 8; ++r) {
      const float mOld = mrow[r];
      const float mNew = fmaxf(mOld, rmax[r]);
      const float alpha = __expf(mOld - mNew);
      const float p0 = __expf(pv0[r] - mNew);
      const float p1 = __expf(pv1[r] - mNew);
      float rs = p0 + p1;
      rs += __shfl_xor(rs, 1, 32);
      rs += __shfl_xor(rs, 2, 32);
      rs += __shfl_xor(rs, 4, 32);
      rs += __shfl_xor(rs, 8, 32);
      lrow[r] = lrow[r] * alpha + rs;
      mrow[r] = mNew;
#pragma unroll
      for (int t = 0; t < 4; ++t) acc[t][r] *= alpha;
      Pw[(r + 8 * hi) * PP + l16]      = (_Float16)p0;
      Pw[(r + 8 * hi) * PP + 16 + l16] = (_Float16)p1;
    }

    asm volatile("s_wait_dscnt 0" ::: "memory");

    {
      const half8 plo = *(const half8*)(&Pw[l16 * PP + hi * 8]);
      const half8 phi = *(const half8*)(&Pw[l16 * PP + hi * 8 + 16]);
      const v16h pA = pack16(plo, phi);
#pragma unroll
      for (int nt = 0; nt < 4; ++nt) {
        const half8 b0 = *(const half8*)(&Vt[(nt * 16 + l16) * VP + hi * 8]);
        const half8 b1 = *(const half8*)(&Vt[(nt * 16 + l16) * VP + hi * 8 + 16]);
        const v16h vB = pack16(b0, b1);
        acc[nt] = __builtin_amdgcn_wmma_f32_16x16x32_f16(false, pA, false, vB,
                                                         (short)0, acc[nt], false, false);
      }
    }
  }

#pragma unroll
  for (int t = 0; t < 4; ++t) {
#pragma unroll
    for (int r = 0; r < 8; ++r) {
      const int qRow = qBase + wave * 16 + r + 8 * hi;
      Oh[(size_t)qRow * DHEAD + t * 16 + l16] = acc[t][r] / lrow[r];
    }
  }
}

// =====================  host launcher  =====================

extern "C" void kernel_launch(void* const* d_in, const int* in_sizes, int n_in,
                              void* d_out, int out_size, void* d_ws, size_t ws_size,
                              hipStream_t stream) {
  (void)n_in; (void)out_size;
  const float* Q = (const float*)d_in[0];
  const float* K = (const float*)d_in[1];
  const float* V = (const float*)d_in[2];
  const unsigned char* M = (const unsigned char*)d_in[3];
  float* O = (float*)d_out;
  const int n = in_sizes[0] / (SEQ * DHEAD);  // heads

  const size_t szK = (size_t)n * SEQ * DHEAD * sizeof(_Float16);
  const size_t szV = szK;
  const size_t szM = (size_t)n * SEQ * (SEQ / 8);  // 1 bit per mask element

  if (d_ws != nullptr && ws_size >= szK + szV + szM) {
    _Float16* Kf = (_Float16*)d_ws;
    _Float16* Vt = (_Float16*)((char*)d_ws + szK);
    unsigned* Mp = (unsigned*)((char*)d_ws + szK + szV);

    const int total8 = n * SEQ * DHEAD / 8;
    hipLaunchKernelGGL(cvt_f16_kernel, dim3((total8 + 255) / 256), dim3(256), 0, stream,
                       K, Kf, total8);
    hipLaunchKernelGGL(tr_v_kernel, dim3(SEQ / 64, n), dim3(256), 0, stream, V, Vt);
    const int totalW = n * SEQ * (SEQ / 32);
    hipLaunchKernelGGL(pack_mask_kernel, dim3((totalW + 255) / 256), dim3(256), 0, stream,
                       M, Mp, totalW);
    hipLaunchKernelGGL(fa_tdm_kernel, dim3(SEQ / BM, n), dim3(128), 0, stream,
                       Q, Kf, Vt, Mp, O);
  } else {
    hipLaunchKernelGGL(fa_fwd_kernel, dim3(SEQ / BM, n), dim3(128), 0, stream,
                       Q, K, V, M, O);
  }
}
